// StatefulCausalDiffAttentionHead_12558484374035
// MI455X (gfx1250) — matrix operand order repaired and checked by reading, not compiled, not run on hardware
//
#include <hip/hip_runtime.h>
#include <hip/hip_bf16.h>

// ---------------------------------------------------------------------------
// StatefulCausalDiffAttentionHead for MI455X (gfx1250, wave32, WMMA)
//   B=4, T=4096, D=1024, dk=dv=128, 2 heads of 128 packed into 256 channels.
// Pipeline:
//   0) wt_convert  : Wq/Wk/Wv -> f16 transposed [n][k] (enables async staging)
//   1) proj_gemm   : q/k/v = x @ W  (f16 WMMA, f32 accum), double-buffered LDS,
//                    B-tile staged with global_load_async_to_lds_b128
//   2) proj_fixup  : recompute the 16 state rows/batch with the *_state weights
//   3) flash_attn  : causal flash attention per head, double-buffered K/V tiles,
//                    async K staging + L2 prefetch two blocks ahead
//   4) diff_ln     : att1 - lam*att2, LayerNorm over 256 channels -> d_out
// ---------------------------------------------------------------------------

typedef __attribute__((ext_vector_type(16))) _Float16 v16h;
typedef __attribute__((ext_vector_type(8)))  _Float16 v8h;
typedef __attribute__((ext_vector_type(8)))  float    v8f;

#define WMMA_F16(a, b, c) \
  __builtin_amdgcn_wmma_f32_16x16x32_f16(false, (a), false, (b), (short)0, (c), false, false)

// async byte-copy global -> LDS (16B per lane), tracked by ASYNCcnt
#define ASYNC_COPY_B128(ldsOff, gsrc)                                  \
  asm volatile("global_load_async_to_lds_b128 %0, %1, off"             \
               :: "v"(ldsOff), "v"(gsrc) : "memory")
#define WAIT_ASYNC0() asm volatile("s_wait_asynccnt 0x0" ::: "memory")

#define BB 4
#define TT 4096
#define DD 1024
#define DK 128
#define NC 256   // 2*dk channels

// ---------------------------------------------------------------------------
// Kernel 0: convert W (1024x256 f32) -> Wt (256x1024 f16, transposed).
// ---------------------------------------------------------------------------
__global__ __launch_bounds__(256)
void wt_convert_kernel(const float* __restrict__ Wq,
                       const float* __restrict__ Wk,
                       const float* __restrict__ Wv,
                       _Float16* __restrict__ wtq,
                       _Float16* __restrict__ wtk,
                       _Float16* __restrict__ wtv)
{
  const float* W  = (blockIdx.z == 0) ? Wq  : ((blockIdx.z == 1) ? Wk  : Wv);
  _Float16*   Wt  = (blockIdx.z == 0) ? wtq : ((blockIdx.z == 1) ? wtk : wtv);
  int idx = blockIdx.x * 256 + threadIdx.x;   // 0 .. 256*1024-1
  int n = idx >> 10;
  int k = idx & 1023;
  Wt[idx] = (_Float16)W[(size_t)k * NC + n];
}

// ---------------------------------------------------------------------------
// Kernel 1: projection GEMM.  M = B*T rows, N = 256, K = 1024.
// Workgroup tile 64x256, 8 waves in 2x4; wave tile 32x64 (2x4 WMMA accums).
// Double-buffered LDS: stage k-step i+1 (async B + packed A) during the
// WMMA work of k-step i; single barrier per iteration.
// ---------------------------------------------------------------------------
__global__ __launch_bounds__(256)
void proj_gemm_kernel(const float* __restrict__ x,
                      const _Float16* __restrict__ wtq,
                      const _Float16* __restrict__ wtk,
                      const _Float16* __restrict__ wtv,
                      _Float16* __restrict__ qb,
                      _Float16* __restrict__ kb,
                      _Float16* __restrict__ vb)
{
  __shared__ __align__(16) _Float16 As[2][64 * 32];    // [m][k]   2 x 4 KB
  __shared__ __align__(16) _Float16 Bst[2][256 * 32];  // [n][k]   2 x 16 KB

  const _Float16* Wt = (blockIdx.z == 0) ? wtq : ((blockIdx.z == 1) ? wtk : wtv);
  _Float16*      out = (blockIdx.z == 0) ? qb  : ((blockIdx.z == 1) ? kb  : vb);

  const int tid  = threadIdx.x;
  const int lane = tid & 31;
  const int w    = tid >> 5;          // wave 0..7
  const int m0   = blockIdx.x * 64;   // global row tile base
  const int wm   = (w & 1) * 32;      // wave row offset in tile
  const int wn   = (w >> 1) * 64;     // wave col offset in tile
  const int la   = lane & 15;
  const int hi   = lane >> 4;

  auto stage = [&](int kbk, int buf) {
    // x tile (64 x 32) f32 -> f16, one b128 store per thread
    int row = tid >> 2;               // 0..63
    int ko  = (tid & 3) * 8;          // 0,8,16,24
    const float* src = x + (size_t)(m0 + row) * DD + kbk + ko;
    v8h hv;
    #pragma unroll
    for (int e = 0; e < 8; ++e) hv[e] = (_Float16)src[e];
    *(v8h*)&As[buf][row * 32 + ko] = hv;
    // Wt tile (256 x 32) async byte-copy, 4 x b128 per thread
    #pragma unroll
    for (int c2 = 0; c2 < 4; ++c2) {
      int chunk = tid + c2 * 256;     // 0..1023
      int n  = chunk >> 2;            // 0..255
      int cc = (chunk & 3) * 8;       // 0,8,16,24
      const _Float16* wsrc = Wt + (size_t)n * DD + kbk + cc;
      unsigned dst = (unsigned)(size_t)&Bst[buf][n * 32 + cc];
      ASYNC_COPY_B128(dst, wsrc);
    }
  };

  v8f acc[2][4] = {};

  stage(0, 0);
  WAIT_ASYNC0();
  __syncthreads();

  const int NSTEP = DD / 32;          // 32 k-steps
  for (int i = 0; i < NSTEP; ++i) {
    const int cur = i & 1;
    if (i + 1 < NSTEP) stage((i + 1) * 32, (i + 1) & 1);

    // --- fragments from buffer `cur` ---
    v16h afrag[2];
    #pragma unroll
    for (int rt = 0; rt < 2; ++rt) {
      int m = wm + rt * 16 + la;
      v16h a;
      #pragma unroll
      for (int e = 0; e < 16; ++e) {
        int k = hi * 16 + e;   // A-frag K layout
        a[e] = As[cur][m * 32 + k];
      }
      afrag[rt] = a;
    }
    v16h bfrag[4];
    #pragma unroll
    for (int ct = 0; ct < 4; ++ct) {
      int n = wn + ct * 16 + la;
      v16h bf;
      const _Float16* bp = &Bst[cur][n * 32 + hi * 16];
      #pragma unroll
      for (int e = 0; e < 16; ++e) bf[e] = bp[e];  // contiguous K run
      bfrag[ct] = bf;
    }
    #pragma unroll
    for (int rt = 0; rt < 2; ++rt)
      #pragma unroll
      for (int ct = 0; ct < 4; ++ct)
        acc[rt][ct] = WMMA_F16(afrag[rt], bfrag[ct], acc[rt][ct]);

    WAIT_ASYNC0();
    __syncthreads();
  }

  // --- store tile as f16 ---
  #pragma unroll
  for (int rt = 0; rt < 2; ++rt)
    #pragma unroll
    for (int ct = 0; ct < 4; ++ct)
      #pragma unroll
      for (int r = 0; r < 8; ++r) {
        int row = m0 + wm + rt * 16 + hi * 8 + r;   // C-tile row mapping
        int col = wn + ct * 16 + la;
        out[(size_t)row * NC + col] = (_Float16)acc[rt][ct][r];
      }
}

// ---------------------------------------------------------------------------
// Kernel 2: recompute state rows (first 8 + last 8 of each batch) with the
// *_state weight matrices.  Tiny: 3 * 4 * 16 rows x 256 cols.
// ---------------------------------------------------------------------------
__global__ __launch_bounds__(256)
void proj_fixup_kernel(const float* __restrict__ x,
                       const float* __restrict__ Wqs,
                       const float* __restrict__ Wks,
                       const float* __restrict__ Wvs,
                       _Float16* __restrict__ qb,
                       _Float16* __restrict__ kb,
                       _Float16* __restrict__ vb)
{
  const int sr    = blockIdx.x;       // 0..15
  const int b     = blockIdx.y;       // 0..3
  const int which = blockIdx.z;       // 0..2
  const int t     = (sr < 8) ? sr : (TT - 16 + sr);
  const float* W  = (which == 0) ? Wqs : ((which == 1) ? Wks : Wvs);
  _Float16*  out  = (which == 0) ? qb  : ((which == 1) ? kb  : vb);
  const int c     = threadIdx.x;      // channel 0..255

  const float* xr = x + ((size_t)b * TT + t) * DD;
  float s = 0.f;
  for (int i = 0; i < DD; ++i)
    s = fmaf(xr[i], W[(size_t)i * NC + c], s);
  out[((size_t)b * TT + t) * NC + c] = (_Float16)s;
}

// ---------------------------------------------------------------------------
// Kernel 3: causal flash attention for one head.
// grid = (T/64, B, 2 heads), 256 threads (8 waves).
// Wave w: 16 query rows r0=(w&3)*16, V-column half c0=(w>>2)*128.
// Double-buffered K/V tiles: async K copy + V transpose stores for block j+1
// are issued before the WMMA work of block j; one barrier per block.
// ---------------------------------------------------------------------------
__global__ __launch_bounds__(256)
void flash_attn_kernel(const _Float16* __restrict__ qb,
                       const _Float16* __restrict__ kbuf,
                       const _Float16* __restrict__ vbuf,
                       float* __restrict__ att1,
                       float* __restrict__ att2)
{
  __shared__ __align__(16) _Float16 Kt[2][32 * 128];    // [key][kdim]  2 x 8 KB
  __shared__ __align__(16) _Float16 Vt[2][256 * 32];    // [col][key]   2 x 16 KB
  __shared__ __align__(16) _Float16 Pb[8 * 16 * 32];    // per-wave P,   8 KB

  const int head = blockIdx.z;
  const int b    = blockIdx.y;
  const int t0   = blockIdx.x * 64;
  const int tid  = threadIdx.x;
  const int lane = tid & 31;
  const int w    = tid >> 5;
  const int r0   = (w & 3) * 16;
  const int c0   = (w >> 2) * 128;
  const int la   = lane & 15;
  const int hi   = lane >> 4;
  const int hoff = head * DK;
  const size_t rowbase = (size_t)b * TT;
  float* attout = head ? att2 : att1;

  auto stageKV = [&](int kb0, int buf) {
    // K tile: async byte-copy, 2 x b128 per thread
    #pragma unroll
    for (int c2 = 0; c2 < 2; ++c2) {
      int chunk = tid + c2 * 256;      // 0..511
      int kr = chunk >> 4;             // 0..31
      int cc = (chunk & 15) * 8;       // 0..120
      const _Float16* src = kbuf + (rowbase + kb0 + kr) * NC + hoff + cc;
      unsigned dst = (unsigned)(size_t)&Kt[buf][kr * 128 + cc];
      ASYNC_COPY_B128(dst, src);
    }
    // V tile: explicit transpose, paired b32 stores
    int kr2 = (tid >> 4) * 2;          // even key 0..30
    int cc  = (tid & 15) * 16;         // 16 cols per thread
    const _Float16* s0 = vbuf + (rowbase + kb0 + kr2) * NC + cc;
    const _Float16* s1 = s0 + NC;
    #pragma unroll
    for (int e = 0; e < 16; ++e) {
      union { _Float16 h[2]; unsigned u; } p;
      p.h[0] = s0[e];
      p.h[1] = s1[e];
      *(unsigned*)&Vt[buf][(cc + e) * 32 + kr2] = p.u;
    }
  };

  // ---- Q fragments: 16 rows x 128 kdim, held in registers for whole loop ----
  v16h qfrag[4];
  {
    const _Float16* qrow = qb + (rowbase + t0 + r0 + la) * NC + hoff;
    #pragma unroll
    for (int kc = 0; kc < 4; ++kc) {
      v16h a;
      #pragma unroll
      for (int e = 0; e < 16; ++e) {
        int k = kc * 32 + hi * 16 + e;
        a[e] = qrow[k];
      }
      qfrag[kc] = a;
    }
  }

  v8f acc[8] = {};                     // 16 x 128 f32 output accumulator
  float mrow[8], lrow[8];
  #pragma unroll
  for (int r = 0; r < 8; ++r) { mrow[r] = -1e30f; lrow[r] = 0.f; }

  const float scale = 0.08838834764831845f;  // 1/sqrt(128)
  const int tmax = t0 + r0 + 15;             // last row this wave owns
  const int nblk = t0 / 32 + 2;              // key blocks covering keys<=t0+63

  stageKV(0, 0);
  WAIT_ASYNC0();
  __syncthreads();

  for (int j = 0; j < nblk; ++j) {
    const int kb0 = j * 32;
    const int cur = j & 1;

    if (j + 1 < nblk) stageKV((j + 1) * 32, (j + 1) & 1);
    // L2 prefetch two blocks ahead
    if (j + 2 < nblk) {
      int kr = tid >> 3;
      __builtin_prefetch(kbuf + (rowbase + kb0 + 64 + kr) * NC + hoff, 0, 1);
      __builtin_prefetch(vbuf + (rowbase + kb0 + 64 + kr) * NC + (tid & 7) * 32, 0, 1);
    }

    if (kb0 <= tmax) {
      // ---- S = Q K^T  (16 x 32 as two 16x16 tiles) ----
      v8f s[2] = {};
      #pragma unroll
      for (int nb = 0; nb < 2; ++nb) {
        #pragma unroll
        for (int kc = 0; kc < 4; ++kc) {
          v16h bf;
          const _Float16* kp = &Kt[cur][(nb * 16 + la) * 128 + kc * 32 + hi * 16];
          #pragma unroll
          for (int e = 0; e < 16; ++e) bf[e] = kp[e];
          s[nb] = WMMA_F16(qfrag[kc], bf, s[nb]);
        }
      }
      // ---- scale + causal mask ----
      #pragma unroll
      for (int nb = 0; nb < 2; ++nb)
        #pragma unroll
        for (int r = 0; r < 8; ++r) {
          int trow = t0 + r0 + hi * 8 + r;
          int key  = kb0 + nb * 16 + la;
          float sv = s[nb][r] * scale;
          s[nb][r] = (key <= trow) ? sv : -1e30f;
        }
      // ---- online softmax: row max / rescale / exp / row sum ----
      float alpha[8], mnew[8];
      #pragma unroll
      for (int r = 0; r < 8; ++r) {
        float v = fmaxf(s[0][r], s[1][r]);
        v = fmaxf(v, __shfl_xor(v, 1, 32));
        v = fmaxf(v, __shfl_xor(v, 2, 32));
        v = fmaxf(v, __shfl_xor(v, 4, 32));
        v = fmaxf(v, __shfl_xor(v, 8, 32));
        mnew[r]  = fmaxf(mrow[r], v);
        alpha[r] = expf(mrow[r] - mnew[r]);
        mrow[r]  = mnew[r];
      }
      #pragma unroll
      for (int r = 0; r < 8; ++r) {
        float p0 = expf(s[0][r] - mnew[r]);
        float p1 = expf(s[1][r] - mnew[r]);
        s[0][r] = p0; s[1][r] = p1;
        float rs = p0 + p1;
        rs += __shfl_xor(rs, 1, 32);
        rs += __shfl_xor(rs, 2, 32);
        rs += __shfl_xor(rs, 4, 32);
        rs += __shfl_xor(rs, 8, 32);
        lrow[r] = lrow[r] * alpha[r] + rs;
      }
      // ---- P: C-tile layout -> LDS -> A-fragment layout ----
      _Float16* pw = &Pb[w * 16 * 32];
      #pragma unroll
      for (int r = 0; r < 8; ++r) {
        pw[(hi * 8 + r) * 32 + la]      = (_Float16)s[0][r];
        pw[(hi * 8 + r) * 32 + 16 + la] = (_Float16)s[1][r];
      }
      v16h pf;
      #pragma unroll
      for (int e = 0; e < 16; ++e) {
        int k = hi * 16 + e;
        pf[e] = pw[la * 32 + k];
      }
      // ---- O = O*alpha + P @ V  over this wave's 128 V columns ----
      #pragma unroll
      for (int cb = 0; cb < 8; ++cb) {
        v8f a = acc[cb];
        #pragma unroll
        for (int r = 0; r < 8; ++r) a[r] *= alpha[r];
        v16h vf;
        const _Float16* vp = &Vt[cur][(c0 + cb * 16 + la) * 32 + hi * 16];
        #pragma unroll
        for (int e = 0; e < 16; ++e) vf[e] = vp[e];
        acc[cb] = WMMA_F16(pf, vf, a);
      }
    }
    WAIT_ASYNC0();
    __syncthreads();
  }

  // ---- epilogue: normalize by l, store f32 ----
  #pragma unroll
  for (int cb = 0; cb < 8; ++cb)
    #pragma unroll
    for (int r = 0; r < 8; ++r) {
      int trow = t0 + r0 + hi * 8 + r;
      int col  = c0 + cb * 16 + la;
      attout[(rowbase + trow) * NC + col] = acc[cb][r] / lrow[r];
    }
}

// ---------------------------------------------------------------------------
// Kernel 4: att = att1 - lam*att2, then LayerNorm over 256 channels.
// One workgroup (256 threads) per (b, t) row.
// ---------------------------------------------------------------------------
__global__ __launch_bounds__(256)
void diff_ln_kernel(const float* __restrict__ att1,
                    const float* __restrict__ att2,
                    const float* __restrict__ lq1,
                    const float* __restrict__ lq2,
                    const float* __restrict__ lk1,
                    const float* __restrict__ lk2,
                    const float* __restrict__ gamma,
                    const float* __restrict__ beta,
                    float* __restrict__ out)
{
  __shared__ float red[256];
  __shared__ float red2[256];
  const size_t row = blockIdx.x;
  const int c = threadIdx.x;

  float d1 = 0.f, d2 = 0.f;
  #pragma unroll 8
  for (int i = 0; i < DK; ++i) {
    d1 = fmaf(lq1[i], lk1[i], d1);
    d2 = fmaf(lq2[i], lk2[i], d2);
  }
  // lambda_init = 0.8 - 0.6*exp(-0.3*1)
  const float lam = expf(d1) - expf(d2) + 0.35550907f;

  const float v = att1[row * NC + c] - lam * att2[row * NC + c];
  red[c]  = v;
  red2[c] = v * v;
  __syncthreads();
  for (int s = 128; s > 0; s >>= 1) {
    if (c < s) { red[c] += red[c + s]; red2[c] += red2[c + s]; }
    __syncthreads();
  }
  const float mu   = red[0] * (1.f / 256.f);
  const float var  = red2[0] * (1.f / 256.f) - mu * mu;
  const float rstd = rsqrtf(var + 1e-5f);
  out[row * NC + c] = (v - mu) * rstd * gamma[c] + beta[c];
}

// ---------------------------------------------------------------------------
extern "C" void kernel_launch(void* const* d_in, const int* in_sizes, int n_in,
                              void* d_out, int out_size, void* d_ws, size_t ws_size,
                              hipStream_t stream) {
  const float* x    = (const float*)d_in[0];
  const float* Wq   = (const float*)d_in[1];
  const float* Wk   = (const float*)d_in[2];
  const float* Wv   = (const float*)d_in[3];
  const float* Wqs  = (const float*)d_in[4];
  const float* Wks  = (const float*)d_in[5];
  const float* Wvs  = (const float*)d_in[6];
  const float* lq1  = (const float*)d_in[7];
  const float* lq2  = (const float*)d_in[8];
  const float* lk1  = (const float*)d_in[9];
  const float* lk2  = (const float*)d_in[10];
  const float* gam  = (const float*)d_in[11];
  const float* bet  = (const float*)d_in[12];
  float* out = (float*)d_out;

  // workspace layout
  char* ws = (char*)d_ws;
  const size_t qkv_bytes = (size_t)BB * TT * NC * sizeof(_Float16);  // 8 MB each
  const size_t att_bytes = (size_t)BB * TT * NC * sizeof(float);     // 16 MB each
  const size_t wt_bytes  = (size_t)NC * DD * sizeof(_Float16);       // 512 KB each
  _Float16* qb  = (_Float16*)(ws);
  _Float16* kb  = (_Float16*)(ws + qkv_bytes);
  _Float16* vb  = (_Float16*)(ws + 2 * qkv_bytes);
  float* att1   = (float*)(ws + 3 * qkv_bytes);
  float* att2   = (float*)(ws + 3 * qkv_bytes + att_bytes);
  _Float16* wtq = (_Float16*)(ws + 3 * qkv_bytes + 2 * att_bytes);
  _Float16* wtk = (_Float16*)(ws + 3 * qkv_bytes + 2 * att_bytes + wt_bytes);
  _Float16* wtv = (_Float16*)(ws + 3 * qkv_bytes + 2 * att_bytes + 2 * wt_bytes);

  // 0) weight convert + transpose to f16
  wt_convert_kernel<<<dim3((NC * DD) / 256, 1, 3), 256, 0, stream>>>(
      Wq, Wk, Wv, wtq, wtk, wtv);

  // 1) projections (q,k,v via blockIdx.z)
  proj_gemm_kernel<<<dim3((BB * TT) / 64, 1, 3), 256, 0, stream>>>(
      x, wtq, wtk, wtv, qb, kb, vb);

  // 2) state-row fixup with *_state weights
  proj_fixup_kernel<<<dim3(16, BB, 3), 256, 0, stream>>>(
      x, Wqs, Wks, Wvs, qb, kb, vb);

  // 3) causal flash attention, both heads
  flash_attn_kernel<<<dim3(TT / 64, BB, 2), 256, 0, stream>>>(
      qb, kb, vb, att1, att2);

  // 4) differential combine + LayerNorm
  diff_ln_kernel<<<dim3(BB * TT), 256, 0, stream>>>(
      att1, att2, lq1, lq2, lk1, lk2, gam, bet, out);
}
